// PathDecoder_14628658610295
// MI455X (gfx1250) — matrix operand order repaired and checked by reading, not compiled, not run on hardware
//
#include <hip/hip_runtime.h>

typedef float  v8f   __attribute__((ext_vector_type(8)));
typedef __bf16 v16bf __attribute__((ext_vector_type(16)));
typedef __bf16 v8bf  __attribute__((ext_vector_type(8)));
typedef __bf16 v4bf  __attribute__((ext_vector_type(4)));

#define B_SZ   256
#define CMAX_  200
#define E_SZ   512
#define V_SZ   27000
#define L_SZ   16

// split fp32 into bf16 hi + bf16 lo (3-term product scheme, ~fp32 accuracy)
__device__ __forceinline__ void bf16split(float v, __bf16& hi, __bf16& lo)
{
    __bf16 h = (__bf16)v;
    hi = h;
    lo = (__bf16)(v - (float)h);
}

__device__ __forceinline__ void split4(const float4 v, v4bf& h4, v4bf& l4)
{
    __bf16 hh, ll;
    bf16split(v.x, hh, ll); h4[0] = hh; l4[0] = ll;
    bf16split(v.y, hh, ll); h4[1] = hh; l4[1] = ll;
    bf16split(v.z, hh, ll); h4[2] = hh; l4[2] = ll;
    bf16split(v.w, hh, ll); h4[3] = hh; l4[3] = ll;
}

// ---------------------------------------------------------------------------
// fp32-in / fp32-out GEMM via bf16x3 split + V_WMMA_F32_16X16X32_BF16.
//   C[M,N] = A[M,K] * B (+bias1+bias2) (+= C if ACCUM)
//   BT=true : logical B[k][n] = Bm[n*ldb + k]  (multiply by Bm^T; needs N%128==0)
//   BT=false: logical B[k][n] = Bm[k*ldb + n]  (N tail handled)
// Block 256 threads (8 waves), block tile 32x128, wave tile 16x32 (2 accs
// sharing one A fragment), BK=32.  Software-pipelined: next K-chunk's global
// loads are issued into registers before the WMMAs of the current chunk, so
// L2 latency overlaps with matrix work.  6 WMMAs per chunk per wave.
// Requires M%32==0, K%32==0 (true at all call sites).
// ---------------------------------------------------------------------------
template <bool BT, bool ACCUM>
__global__ __launch_bounds__(256) void wmma_gemm_bf16x3(
    const float* __restrict__ A, int lda,
    const float* __restrict__ Bm, int ldb,
    float* __restrict__ C, int ldc,
    int M, int N, int K,
    const float* __restrict__ bias1,
    const float* __restrict__ bias2)
{
    constexpr int BM = 32, BN = 128, BK = 32;
    constexpr int AP = 40;   // bf16 pitch: keeps 8-elt K-runs 16B aligned
    constexpr int BP = 40;
    __shared__ __bf16 Ah[BM][AP], Al[BM][AP];
    __shared__ __bf16 Bh[BN][BP], Bl[BN][BP];   // transposed: [n][k]

    const int tid  = threadIdx.x;
    const int wid  = tid >> 5;
    const int lane = tid & 31;
    const int half = lane >> 4;
    const int l16  = lane & 15;
    const int m0   = blockIdx.y * BM;
    const int n0   = blockIdx.x * BN;
    const int mw   = (wid >> 2) * 16;       // wave row offset (2 rows of waves)
    const int nw   = (wid & 3) * 32;        // wave col offset (4 cols, 32 wide)

    const int ar = tid >> 3, ak4 = (tid & 7) * 4;   // A staging coords

    v8f acc0 = {0.f, 0.f, 0.f, 0.f, 0.f, 0.f, 0.f, 0.f};
    v8f acc1 = {0.f, 0.f, 0.f, 0.f, 0.f, 0.f, 0.f, 0.f};

    float4 a_reg;
    float4 b_reg[4];

    // issue global loads of one K-chunk into registers (+ prefetch next)
    auto load_tiles = [&](int k0) {
        const float* asrc = A + (size_t)(m0 + ar) * lda + k0 + ak4;
        a_reg = *(const float4*)asrc;
        if (k0 + BK < K) __builtin_prefetch(asrc + BK, 0, 3);
        if (BT) {
            #pragma unroll
            for (int j = 0; j < 4; ++j) {
                int q = tid + j * 256;
                int n = q >> 3, k4 = (q & 7) * 4;
                const float* src = Bm + (size_t)(n0 + n) * ldb + k0 + k4;
                b_reg[j] = *(const float4*)src;
                if (k0 + BK < K) __builtin_prefetch(src + BK, 0, 3);
            }
        } else {
            #pragma unroll
            for (int j = 0; j < 4; ++j) {
                int q = tid + j * 256;
                int kk = q >> 5, n4 = (q & 31) * 4;
                const float* src = Bm + (size_t)(k0 + kk) * ldb + n0 + n4;
                if (n0 + n4 + 3 < N) {
                    b_reg[j] = *(const float4*)src;
                } else {
                    b_reg[j] = make_float4(0.f, 0.f, 0.f, 0.f);
                    if (n0 + n4 + 0 < N) b_reg[j].x = src[0];
                    if (n0 + n4 + 1 < N) b_reg[j].y = src[1];
                    if (n0 + n4 + 2 < N) b_reg[j].z = src[2];
                }
                if (k0 + BK < K) __builtin_prefetch(src + (size_t)BK * ldb, 0, 3);
            }
        }
    };

    load_tiles(0);

    for (int k0 = 0; k0 < K; k0 += BK) {
        // ---- convert current registers to bf16 hi/lo and store to LDS ----
        {
            v4bf h4, l4;
            split4(a_reg, h4, l4);
            *(v4bf*)&Ah[ar][ak4] = h4;
            *(v4bf*)&Al[ar][ak4] = l4;
        }
        if (BT) {
            #pragma unroll
            for (int j = 0; j < 4; ++j) {
                int q = tid + j * 256;
                int n = q >> 3, k4 = (q & 7) * 4;
                v4bf h4, l4;
                split4(b_reg[j], h4, l4);
                *(v4bf*)&Bh[n][k4] = h4;
                *(v4bf*)&Bl[n][k4] = l4;
            }
        } else {
            #pragma unroll
            for (int j = 0; j < 4; ++j) {
                int q = tid + j * 256;
                int kk = q >> 5, n4 = (q & 31) * 4;
                __bf16 hh, ll;
                bf16split(b_reg[j].x, hh, ll); Bh[n4 + 0][kk] = hh; Bl[n4 + 0][kk] = ll;
                bf16split(b_reg[j].y, hh, ll); Bh[n4 + 1][kk] = hh; Bl[n4 + 1][kk] = ll;
                bf16split(b_reg[j].z, hh, ll); Bh[n4 + 2][kk] = hh; Bl[n4 + 2][kk] = ll;
                bf16split(b_reg[j].w, hh, ll); Bh[n4 + 3][kk] = hh; Bl[n4 + 3][kk] = ll;
            }
        }
        __syncthreads();

        // ---- issue next chunk's global loads; waits land at next convert ----
        if (k0 + BK < K) load_tiles(k0 + BK);

        // ---- A fragments (shared by both accumulators) ----
        // A 16x32 layout: elem i<8 -> K = half*8+i ; i>=8 -> K = 16+half*8+(i-8)
        const __bf16* arh = &Ah[mw + l16][0];
        const __bf16* arl = &Al[mw + l16][0];
        v8bf t0 = *(const v8bf*)(arh + half * 8);
        v8bf t1 = *(const v8bf*)(arh + 16 + half * 8);
        v16bf ah = __builtin_shufflevector(t0, t1, 0,1,2,3,4,5,6,7,8,9,10,11,12,13,14,15);
        t0 = *(const v8bf*)(arl + half * 8);
        t1 = *(const v8bf*)(arl + 16 + half * 8);
        v16bf al = __builtin_shufflevector(t0, t1, 0,1,2,3,4,5,6,7,8,9,10,11,12,13,14,15);

        // ---- B fragments + WMMAs, two 16-wide column halves ----
        // B 32x16 layout: elem i<8 -> K = half*8+i ; i>=8 -> K = 16+half*8+(i-8) at N = l16 (stored transposed)
        #pragma unroll
        for (int nh = 0; nh < 2; ++nh) {
            const __bf16* brh = &Bh[nw + nh * 16 + l16][0];
            const __bf16* brl = &Bl[nw + nh * 16 + l16][0];
            t0 = *(const v8bf*)(brh + half * 8);
            t1 = *(const v8bf*)(brh + 16 + half * 8);
            v16bf bh = __builtin_shufflevector(t0, t1, 0,1,2,3,4,5,6,7,8,9,10,11,12,13,14,15);
            t0 = *(const v8bf*)(brl + half * 8);
            t1 = *(const v8bf*)(brl + 16 + half * 8);
            v16bf bl = __builtin_shufflevector(t0, t1, 0,1,2,3,4,5,6,7,8,9,10,11,12,13,14,15);
            v8f acc = nh ? acc1 : acc0;
            // small cross terms first, then the dominant hi*hi term
            acc = __builtin_amdgcn_wmma_f32_16x16x32_bf16(false, ah, false, bl, (short)0, acc, false, false);
            acc = __builtin_amdgcn_wmma_f32_16x16x32_bf16(false, al, false, bh, (short)0, acc, false, false);
            acc = __builtin_amdgcn_wmma_f32_16x16x32_bf16(false, ah, false, bh, (short)0, acc, false, false);
            if (nh) acc1 = acc; else acc0 = acc;
        }
        __syncthreads();
    }

    // ---- epilogue: f32 C/D layout: VGPR r -> row r+8*half, col l16 ----
    #pragma unroll
    for (int nh = 0; nh < 2; ++nh) {
        const int col = n0 + nw + nh * 16 + l16;
        if (col < N) {
            float badd = 0.f;
            if (bias1) badd += bias1[col];
            if (bias2) badd += bias2[col];
            const v8f& acc = nh ? acc1 : acc0;
            #pragma unroll
            for (int r = 0; r < 8; ++r) {
                int row = m0 + mw + r + 8 * half;
                size_t off = (size_t)row * ldc + col;
                float v = acc[r] + badd;
                if (ACCUM) v += C[off];
                C[off] = v;
            }
        }
    }
}

// ---------------------------------------------------------------------------
// Exclusive scan of counts[256] -> offsets[256]  (1 block, 256 threads)
// ---------------------------------------------------------------------------
__global__ void scan_kernel(const int* __restrict__ counts, int* __restrict__ offsets)
{
    __shared__ int s[B_SZ];
    int tid = threadIdx.x;
    int v = counts[tid];
    s[tid] = v;
    __syncthreads();
    for (int st = 1; st < B_SZ; st <<= 1) {
        int add = (tid >= st) ? s[tid - st] : 0;
        __syncthreads();
        s[tid] += add;
        __syncthreads();
    }
    offsets[tid] = s[tid] - v;
}

// h0 = c0 = mean over real contexts (ragged read of encoded_paths)
__global__ __launch_bounds__(256) void init_state_kernel(
    const float* __restrict__ enc, const int* __restrict__ offsets,
    const int* __restrict__ counts, float* __restrict__ h, float* __restrict__ c)
{
    int b = blockIdx.x, tid = threadIdx.x;
    int off = offsets[b], cnt = counts[b];
    float inv = 1.f / (float)cnt;
    for (int e = tid; e < E_SZ; e += 256) {
        float s = 0.f;
        for (int k = 0; k < cnt; ++k)
            s += enc[(size_t)(off + k) * E_SZ + e];
        float v = s * inv;
        h[(size_t)b * E_SZ + e] = v;
        c[(size_t)b * E_SZ + e] = v;
    }
}

__global__ void cur_init_kernel(const int* __restrict__ tl, int* __restrict__ cur)
{
    cur[threadIdx.x] = tl[threadIdx.x];   // target_labels[0, :]
}

__global__ void zero_kernel(float* __restrict__ p, long long n)
{
    long long i = (long long)blockIdx.x * 256 + threadIdx.x;
    if (i < n) p[i] = 0.f;
}

// ---------------------------------------------------------------------------
// Attention: aw[c] = ctx[b,c,:].g[b,:] + attn_b.h[b,:]; softmax over valid c;
// attended[b,:] = sum_c score[c]*ctx[b,c,:]       (one block per label b)
// ---------------------------------------------------------------------------
__global__ __launch_bounds__(256) void attention_kernel(
    const float* __restrict__ enc, const int* __restrict__ offsets,
    const int* __restrict__ counts, const float* __restrict__ g,
    const float* __restrict__ h, const float* __restrict__ attn_b,
    float* __restrict__ attended)
{
    const int b = blockIdx.x;
    const int tid = threadIdx.x;
    const int cnt = counts[b];
    const int off = offsets[b];
    __shared__ float gsh[E_SZ];
    __shared__ float aw[CMAX_];
    __shared__ float red[256];

    // cache g[b] in LDS; sb = attn_b . h[b]
    float s = 0.f;
    for (int e = tid; e < E_SZ; e += 256) {
        gsh[e] = g[(size_t)b * E_SZ + e];
        s += attn_b[e] * h[(size_t)b * E_SZ + e];
    }
    red[tid] = s; __syncthreads();
    for (int st = 128; st > 0; st >>= 1) { if (tid < st) red[tid] += red[tid + st]; __syncthreads(); }
    const float sb = red[0];
    __syncthreads();

    // one context row per wave iteration: float4 512-dot across 32 lanes
    const int wid = tid >> 5, lane = tid & 31;
    const float4* g4 = (const float4*)gsh;
    for (int cc = wid; cc < cnt; cc += 8) {
        const float4* row4 = (const float4*)(enc + (size_t)(off + cc) * E_SZ);
        float d = 0.f;
        #pragma unroll
        for (int q = lane; q < E_SZ / 4; q += 32) {
            float4 r = row4[q], gg = g4[q];
            d += r.x * gg.x + r.y * gg.y + r.z * gg.z + r.w * gg.w;
        }
        for (int m = 16; m > 0; m >>= 1) d += __shfl_xor(d, m, 32);
        if (lane == 0) aw[cc] = d + sb;
    }
    __syncthreads();

    // softmax over aw[0..cnt)   (== NEG masking of invalid entries)
    float mx = -3.4e38f;
    for (int cc = tid; cc < cnt; cc += 256) mx = fmaxf(mx, aw[cc]);
    red[tid] = mx; __syncthreads();
    for (int st = 128; st > 0; st >>= 1) { if (tid < st) red[tid] = fmaxf(red[tid], red[tid + st]); __syncthreads(); }
    mx = red[0]; __syncthreads();
    float sum = 0.f;
    for (int cc = tid; cc < cnt; cc += 256) { float e = expf(aw[cc] - mx); aw[cc] = e; sum += e; }
    red[tid] = sum; __syncthreads();
    for (int st = 128; st > 0; st >>= 1) { if (tid < st) red[tid] += red[tid + st]; __syncthreads(); }
    const float inv = 1.f / red[0];
    __syncthreads();

    // attended[b,e] = sum_c score[c] * enc[off+c, e]  (coalesced across tid)
    for (int e = tid; e < E_SZ; e += 256) {
        float acc = 0.f;
        for (int cc = 0; cc < cnt; ++cc)
            acc += aw[cc] * enc[(size_t)(off + cc) * E_SZ + e];
        attended[(size_t)b * E_SZ + e] = acc * inv;
    }
}

// x[b] = concat(embedding[cur[b]], attended[b])   (block per b)
__global__ __launch_bounds__(256) void concat_x_kernel(
    const float* __restrict__ emb, const int* __restrict__ cur,
    const float* __restrict__ attended, float* __restrict__ x)
{
    int b = blockIdx.x, tid = threadIdx.x;
    const float* er = emb + (size_t)cur[b] * E_SZ;
    for (int j = tid; j < 2 * E_SZ; j += 256)
        x[(size_t)b * (2 * E_SZ) + j] =
            (j < E_SZ) ? er[j] : attended[(size_t)b * E_SZ + (j - E_SZ)];
}

// LSTM cell pointwise: gate order i,f,g,o (chunks of E)
__global__ void lstm_kernel(const float* __restrict__ gates,
                            float* __restrict__ h, float* __restrict__ c)
{
    int i = blockIdx.x * blockDim.x + threadIdx.x;
    if (i >= B_SZ * E_SZ) return;
    int b = i / E_SZ, e = i % E_SZ;
    const float* gr = gates + (size_t)b * 4 * E_SZ;
    float ig = gr[e], fg = gr[E_SZ + e], gg = gr[2 * E_SZ + e], og = gr[3 * E_SZ + e];
    float si = 1.f / (1.f + expf(-ig));
    float sf = 1.f / (1.f + expf(-fg));
    float so = 1.f / (1.f + expf(-og));
    float cn = sf * c[i] + si * tanhf(gg);
    float hn = so * tanhf(cn);
    c[i] = cn; h[i] = hn;
}

// Greedy argmax over V=27000 (first max index on ties), block per b
__global__ __launch_bounds__(256) void argmax_kernel(
    const float* __restrict__ logits, int* __restrict__ cur)
{
    int b = blockIdx.x, tid = threadIdx.x;
    const float* row = logits + (size_t)b * V_SZ;
    float best = -3.4e38f; int bi = 0x7fffffff;
    for (int v = tid; v < V_SZ; v += 256) {
        float val = row[v];
        if (val > best) { best = val; bi = v; }
    }
    __shared__ float bv[256]; __shared__ int bix[256];
    bv[tid] = best; bix[tid] = bi; __syncthreads();
    for (int st = 128; st > 0; st >>= 1) {
        if (tid < st) {
            if (bv[tid + st] > bv[tid] ||
                (bv[tid + st] == bv[tid] && bix[tid + st] < bix[tid])) {
                bv[tid] = bv[tid + st]; bix[tid] = bix[tid + st];
            }
        }
        __syncthreads();
    }
    if (tid == 0) cur[b] = bix[0];
}

// ---------------------------------------------------------------------------
extern "C" void kernel_launch(void* const* d_in, const int* in_sizes, int n_in,
                              void* d_out, int out_size, void* d_ws, size_t ws_size,
                              hipStream_t stream)
{
    const float* enc    = (const float*)d_in[0];
    const int*   tl     = (const int*)  d_in[1];
    const int*   counts = (const int*)  d_in[2];
    const float* emb    = (const float*)d_in[3];
    const float* attn_W = (const float*)d_in[4];
    const float* attn_b = (const float*)d_in[5];
    const float* W_ih   = (const float*)d_in[6];   // [2048, 1024]
    const float* W_hh   = (const float*)d_in[7];   // [2048, 512]
    const float* b_ih   = (const float*)d_in[8];
    const float* b_hh   = (const float*)d_in[9];
    const float* proj_W = (const float*)d_in[10];  // [512, 27000]
    const float* proj_b = (const float*)d_in[11];
    float* out = (float*)d_out;

    // workspace carve-out (~5.1 MB)
    char* ws = (char*)d_ws;
    size_t p = 0;
    auto alloc = [&](size_t bytes) -> char* {
        char* q = ws + p; p += (bytes + 255) & ~(size_t)255; return q;
    };
    int*   offsets  = (int*)  alloc(B_SZ * sizeof(int));
    int*   cur      = (int*)  alloc(B_SZ * sizeof(int));
    float* h        = (float*)alloc((size_t)B_SZ * E_SZ * sizeof(float));
    float* c        = (float*)alloc((size_t)B_SZ * E_SZ * sizeof(float));
    float* g        = (float*)alloc((size_t)B_SZ * E_SZ * sizeof(float));
    float* attended = (float*)alloc((size_t)B_SZ * E_SZ * sizeof(float));
    float* x        = (float*)alloc((size_t)B_SZ * 2 * E_SZ * sizeof(float));
    float* gates    = (float*)alloc((size_t)B_SZ * 4 * E_SZ * sizeof(float));
    (void)ws_size; (void)in_sizes; (void)n_in; (void)out_size;

    // setup
    scan_kernel<<<1, B_SZ, 0, stream>>>(counts, offsets);
    init_state_kernel<<<B_SZ, 256, 0, stream>>>(enc, offsets, counts, h, c);
    cur_init_kernel<<<1, B_SZ, 0, stream>>>(tl, cur);
    {
        long long n0 = (long long)B_SZ * V_SZ;           // out[0] = 0
        zero_kernel<<<(unsigned)((n0 + 255) / 256), 256, 0, stream>>>(out, n0);
    }

    dim3 blk(256);
    dim3 grid_g((E_SZ + 127) / 128, B_SZ / 32);          // 4 x 8
    dim3 grid_gate((4 * E_SZ + 127) / 128, B_SZ / 32);   // 16 x 8
    dim3 grid_proj((V_SZ + 127) / 128, B_SZ / 32);       // 211 x 8

    for (int t = 1; t < L_SZ; ++t) {
        // g = h @ attn_W^T  (folds attn_W into the per-step h transform)
        wmma_gemm_bf16x3<true, false><<<grid_g, blk, 0, stream>>>(
            h, E_SZ, attn_W, E_SZ, g, E_SZ, B_SZ, E_SZ, E_SZ, nullptr, nullptr);

        // scores + softmax + weighted context (ragged, one block per label)
        attention_kernel<<<B_SZ, 256, 0, stream>>>(
            enc, offsets, counts, g, h, attn_b, attended);

        // x = [embedding[cur], attended]
        concat_x_kernel<<<B_SZ, 256, 0, stream>>>(emb, cur, attended, x);

        // gates = x @ W_ih^T + b_ih + b_hh
        wmma_gemm_bf16x3<true, false><<<grid_gate, blk, 0, stream>>>(
            x, 2 * E_SZ, W_ih, 2 * E_SZ, gates, 4 * E_SZ,
            B_SZ, 4 * E_SZ, 2 * E_SZ, b_ih, b_hh);

        // gates += h @ W_hh^T
        wmma_gemm_bf16x3<true, true><<<grid_gate, blk, 0, stream>>>(
            h, E_SZ, W_hh, E_SZ, gates, 4 * E_SZ,
            B_SZ, 4 * E_SZ, E_SZ, nullptr, nullptr);

        // LSTM pointwise; updates h,c in place
        lstm_kernel<<<(B_SZ * E_SZ + 255) / 256, 256, 0, stream>>>(gates, h, c);

        // logits -> out[t]
        float* out_t = out + (size_t)t * B_SZ * V_SZ;
        wmma_gemm_bf16x3<false, false><<<grid_proj, blk, 0, stream>>>(
            h, E_SZ, proj_W, V_SZ, out_t, V_SZ,
            B_SZ, V_SZ, E_SZ, proj_b, nullptr);

        // greedy feedback (not needed after the final step)
        if (t < L_SZ - 1)
            argmax_kernel<<<B_SZ, 256, 0, stream>>>(out_t, cur);
    }
}
